// QuerySpecificAttentionClusteringModel_3367254360581
// MI455X (gfx1250) — matrix operand order repaired and checked by reading, not compiled, not run on hardware
//
#include <hip/hip_runtime.h>
#include <hip/hip_bf16.h>
#include <math.h>

// ---------------------------------------------------------------------------
// CDNA5 (gfx1250) wave32 WMMA types
// ---------------------------------------------------------------------------
typedef __attribute__((ext_vector_type(16))) __bf16 v16bf;
typedef __attribute__((ext_vector_type(8)))  float  v8f;

union AFrag {                // 16 bf16 per lane = 32 bytes = 8 VGPRs
    v16bf v;
    uint4 q[2];
    unsigned short h[16];
};

__device__ __forceinline__ unsigned short f2bf(float f) {
    unsigned u = __float_as_uint(f);
    unsigned r = u + 0x7FFFu + ((u >> 16) & 1u);   // RNE
    return (unsigned short)(r >> 16);
}

// gfx1250 async global->LDS copy (16B per lane), tracked on ASYNCcnt
__device__ __forceinline__ void async_copy16(unsigned lds_off, const void* gptr) {
    unsigned long long ga = (unsigned long long)(uintptr_t)gptr;
    asm volatile("global_load_async_to_lds_b128 %0, %1, off"
                 :: "v"(lds_off), "v"(ga) : "memory");
}
__device__ __forceinline__ void wait_asynccnt0() {
    asm volatile("s_wait_asynccnt 0x0" ::: "memory");
}

// ---------------------------------------------------------------------------
// One-time packing: f32 -> bf16 (straight and transposed)
// ---------------------------------------------------------------------------
__global__ __launch_bounds__(256) void convert_bf16(
    const float* __restrict__ src, unsigned short* __restrict__ dst, int n) {
    int i = blockIdx.x * 256 + threadIdx.x;
    if (i < n) dst[i] = f2bf(src[i]);
}

// src [R][Cc] f32  ->  dst [Cc][R] bf16
__global__ __launch_bounds__(256) void convert_bf16_T(
    const float* __restrict__ src, unsigned short* __restrict__ dst,
    int R, int Cc) {
    int i = blockIdx.x * 256 + threadIdx.x;
    if (i < R * Cc) {
        int r = i / Cc, c = i % Cc;
        dst[(size_t)c * R + r] = f2bf(src[i]);
    }
}

// ---------------------------------------------------------------------------
// Unified GEMM: acc = A[M,K](bf16) @ Bt[N,K](bf16)^T + bias, f32 accumulate.
// A and B tiles are pure byte copies -> double-buffered async LDS staging.
// Epilogue: C (f32, optional) and/or Cb (bf16, optional):
//   cbmode 1: Cb[row*N+col]          (row-major bf16)
//   cbmode 2: Cb[(col<<12)+row]      (V pre-transpose: Vt[h][d][key], M=4096)
// WG 256 thr = 8 waves; WG tile 128x128; wave tile 64x32 (4x2 frags).
// ---------------------------------------------------------------------------
#define GLDP 40   // LDS pitch in bf16 elems; 80B row stride (16B aligned)

__global__ __launch_bounds__(256) void gemm_bf16_async(
    const unsigned short* __restrict__ A,   // [M,K] bf16
    const unsigned short* __restrict__ Bt,  // [N,K] bf16
    const float* __restrict__ bias,
    float* __restrict__ C, unsigned short* __restrict__ Cb, int cbmode,
    int M, int N, int K)
{
    __shared__ unsigned short As[2][128 * GLDP];
    __shared__ unsigned short Bs[2][128 * GLDP];

    const int t    = threadIdx.x;
    const int lane = t & 31;
    const int wave = t >> 5;
    const int hi   = (lane >> 4) & 1;
    const int ln   = lane & 15;
    const int bm   = blockIdx.x * 128;
    const int bn   = blockIdx.y * 128;
    const int wm   = (wave >> 2) * 64;   // 2 waves along M
    const int wn   = (wave & 3) * 32;    // 4 waves along N

    const unsigned asbase = (unsigned)(uintptr_t)&As[0][0];
    const unsigned bsbase = (unsigned)(uintptr_t)&Bs[0][0];
    const unsigned bufstride = 128 * GLDP * 2;   // bytes

    // stage one 128x32 bf16 tile of A and of Bt into buffer `buf`
    // 512 16B-chunks each; 2 of each per thread.
    auto stage = [&](int k0, int buf) {
        #pragma unroll
        for (int i = 0; i < 2; i++) {
            int id = t + i * 256;                 // 0..511
            int r = id >> 2, c8 = (id & 3) * 8;
            unsigned off = buf * bufstride + (unsigned)(r * GLDP + c8) * 2;
            async_copy16(asbase + off, A  + (size_t)(bm + r) * K + k0 + c8);
            async_copy16(bsbase + off, Bt + (size_t)(bn + r) * K + k0 + c8);
        }
    };

    v8f acc[4][2] = {};

    stage(0, 0);
    wait_asynccnt0();
    __syncthreads();

    const int nk = K >> 5;
    for (int kk = 0; kk < nk; kk++) {
        const int buf = kk & 1;
        if (kk + 1 < nk) stage((kk + 1) * 32, buf ^ 1);   // prefetch next

        // --- B fragments: per lane 2x16B (col=ln, K=hi*8..+7 and 16+hi*8..+7) ---
        AFrag bfr[2];
        #pragma unroll
        for (int fn = 0; fn < 2; fn++) {
            const uint4* p =
                (const uint4*)&Bs[buf][(wn + fn * 16 + ln) * GLDP + hi * 8];
            bfr[fn].q[0] = p[0]; bfr[fn].q[1] = p[2];
        }
        // --- A fragments + WMMA ---
        #pragma unroll
        for (int fm = 0; fm < 4; fm++) {
            AFrag af;
            const unsigned short* base =
                &As[buf][(wm + fm * 16 + ln) * GLDP + hi * 8];
            af.q[0] = *(const uint4*)(base);
            af.q[1] = *(const uint4*)(base + 16);
            #pragma unroll
            for (int fn = 0; fn < 2; fn++) {
                acc[fm][fn] = __builtin_amdgcn_wmma_f32_16x16x32_bf16(
                    false, af.v, false, bfr[fn].v, (short)0, acc[fm][fn],
                    false, false);
            }
        }

        wait_asynccnt0();
        __syncthreads();
    }

    // --- epilogue: C layout lane=(col ln), VGPR r = row r+8*hi ---
    #pragma unroll
    for (int fm = 0; fm < 4; fm++) {
        #pragma unroll
        for (int fn = 0; fn < 2; fn++) {
            int col = bn + wn + fn * 16 + ln;
            float bv = bias ? bias[col] : 0.0f;
            #pragma unroll
            for (int r = 0; r < 8; r++) {
                int row = bm + wm + fm * 16 + hi * 8 + r;
                float val = acc[fm][fn][r] + bv;
                if (C)  C[(size_t)row * N + col] = val;
                if (Cb) {
                    if (cbmode == 1)
                        Cb[(size_t)row * N + col] = f2bf(val);
                    else
                        Cb[((size_t)col << 12) + row] = f2bf(val);
                }
            }
        }
    }
}

// ---------------------------------------------------------------------------
// Flash attention, 8 heads, N=4096, hd=64, scale=1/8.
// bf16 WMMA, f32 online softmax. K/V tiles double-buffered in LDS via
// global_load_async_to_lds_b128 (ASYNCcnt), overlapped with WMMA compute.
// Grid (32 q-tiles, 8 heads), block 256 = 8 waves; wave owns 16 query rows.
// Output written as bf16 row-major (feeds out-projection GEMM).
// ---------------------------------------------------------------------------
#define KP 72   // LDS pitch in bf16 (144B rows, 16B aligned)

__global__ __launch_bounds__(256) void flash_attn(
    const unsigned short* __restrict__ Qb, const unsigned short* __restrict__ Kb,
    const unsigned short* __restrict__ Vtb, unsigned short* __restrict__ Ob)
{
    __shared__ unsigned short Ks[2][64 * KP];    // [key][hd]
    __shared__ unsigned short Vs[2][64 * KP];    // [hd][key] (pre-transposed)
    __shared__ unsigned short Ps[8 * 16 * KP];   // per-wave P [16 q][64 key]

    const int t    = threadIdx.x;
    const int lane = t & 31;
    const int wave = t >> 5;
    const int hi   = (lane >> 4) & 1;
    const int ln   = lane & 15;
    const int h    = blockIdx.y;
    const int hc   = h * 64;
    const int q0   = blockIdx.x * 128 + wave * 16;

    const unsigned ksbase = (unsigned)(uintptr_t)&Ks[0][0];
    const unsigned vsbase = (unsigned)(uintptr_t)&Vs[0][0];
    const unsigned bufstride = 64 * KP * 2;      // bytes per buffer

    // Preload Q A-fragments straight from bf16 global (16B aligned chunks)
    AFrag qa[2];
    {
        const unsigned short* qrow = Qb + (size_t)(q0 + ln) * 512 + hc;
        #pragma unroll
        for (int ks = 0; ks < 2; ks++) {
            const unsigned short* b = qrow + ks * 32 + hi * 8;
            qa[ks].q[0] = *(const uint4*)(b);
            qa[ks].q[1] = *(const uint4*)(b + 16);
        }
    }

    v8f oacc[4] = {};
    float mrun[8], lrun[8];
    #pragma unroll
    for (int r = 0; r < 8; r++) { mrun[r] = -1e30f; lrun[r] = 0.0f; }

    unsigned short* myPs = &Ps[wave * 16 * KP];

    // stage key-block j0 (64 keys x 64 hd, bf16) into LDS buffer `buf`
    const unsigned short* VtbH = Vtb + ((size_t)hc << 12);
    auto stage = [&](int j0, int buf) {
        #pragma unroll
        for (int i = 0; i < 2; i++) {
            int id = t + i * 256;                // 0..511
            int r = id >> 3, c8 = (id & 7) * 8;
            async_copy16(ksbase + buf * bufstride + (unsigned)(r * KP + c8) * 2,
                         Kb + (size_t)(j0 + r) * 512 + hc + c8);
            async_copy16(vsbase + buf * bufstride + (unsigned)(r * KP + c8) * 2,
                         VtbH + ((size_t)r << 12) + j0 + c8);
        }
    };

    stage(0, 0);
    wait_asynccnt0();
    __syncthreads();

    for (int j0 = 0; j0 < 4096; j0 += 64) {
        const int buf = (j0 >> 6) & 1;
        if (j0 + 64 < 4096) stage(j0 + 64, buf ^ 1);   // prefetch next block

        // ---- S = Q K^T  (16 x 64) ----
        v8f sacc[4] = {};
        #pragma unroll
        for (int fn = 0; fn < 4; fn++) {
            #pragma unroll
            for (int ks = 0; ks < 2; ks++) {
                AFrag bf_;
                const uint4* p =
                    (const uint4*)&Ks[buf][(fn * 16 + ln) * KP + ks * 32 + hi * 8];
                bf_.q[0] = p[0]; bf_.q[1] = p[2];
                sacc[fn] = __builtin_amdgcn_wmma_f32_16x16x32_bf16(
                    false, qa[ks].v, false, bf_.v, (short)0, sacc[fn],
                    false, false);
            }
        }

        // ---- online softmax, write P (bf16) to per-wave LDS ----
        #pragma unroll
        for (int r = 0; r < 8; r++) {
            float s[4], mx = -1e30f;
            #pragma unroll
            for (int fn = 0; fn < 4; fn++) {
                s[fn] = sacc[fn][r] * 0.125f;
                mx = fmaxf(mx, s[fn]);
            }
            #pragma unroll
            for (int off = 8; off >= 1; off >>= 1)
                mx = fmaxf(mx, __shfl_xor(mx, off, 16));
            float newm  = fmaxf(mrun[r], mx);
            float alpha = __expf(mrun[r] - newm);
            float ls = 0.0f;
            #pragma unroll
            for (int fn = 0; fn < 4; fn++) {
                float p = __expf(s[fn] - newm);
                ls += p;
                myPs[(r + 8 * hi) * KP + fn * 16 + ln] = f2bf(p);
            }
            #pragma unroll
            for (int off = 8; off >= 1; off >>= 1)
                ls += __shfl_xor(ls, off, 16);
            mrun[r] = newm;
            lrun[r] = lrun[r] * alpha + ls;
            #pragma unroll
            for (int fn = 0; fn < 4; fn++) oacc[fn][r] *= alpha;
        }

        // ---- O += P V ----
        #pragma unroll
        for (int ks = 0; ks < 2; ks++) {
            AFrag pa;
            const unsigned short* b = &myPs[ln * KP + ks * 32 + hi * 8];
            pa.q[0] = *(const uint4*)(b);
            pa.q[1] = *(const uint4*)(b + 16);
            #pragma unroll
            for (int fn = 0; fn < 4; fn++) {
                AFrag vf;
                const uint4* p =
                    (const uint4*)&Vs[buf][(fn * 16 + ln) * KP + ks * 32 + hi * 8];
                vf.q[0] = p[0]; vf.q[1] = p[2];
                oacc[fn] = __builtin_amdgcn_wmma_f32_16x16x32_bf16(
                    false, pa.v, false, vf.v, (short)0, oacc[fn],
                    false, false);
            }
        }

        wait_asynccnt0();      // next buffer landed
        __syncthreads();       // all waves done reading `buf` + see new data
    }

    #pragma unroll
    for (int r = 0; r < 8; r++) {
        float inv = 1.0f / lrun[r];
        int row = q0 + 8 * hi + r;
        #pragma unroll
        for (int fn = 0; fn < 4; fn++)
            Ob[(size_t)row * 512 + hc + fn * 16 + ln] = f2bf(oacc[fn][r] * inv);
    }
}

// ---------------------------------------------------------------------------
// DKM (fp32, flag-gated so the fixed 101-iteration graph matches lax.scan)
// ---------------------------------------------------------------------------
__global__ void xsq_kernel(const float* __restrict__ X, float* __restrict__ Xsq) {
    int i = blockIdx.x * 256 + threadIdx.x;
    float s = 0.0f;
    for (int k = 0; k < 512; k++) { float v = X[(size_t)i * 512 + k]; s += v * v; }
    Xsq[i] = s;
}

__global__ void dkm_init(const float* __restrict__ X, const int* __restrict__ idx,
                         float* __restrict__ C, float* __restrict__ Ct,
                         float* __restrict__ Csq, int* __restrict__ flag) {
    for (int p = threadIdx.x; p < 32 * 512; p += 256) {
        int c = p >> 9, e = p & 511;
        float v = X[(size_t)idx[c] * 512 + e];
        C[p] = v; Ct[e * 32 + c] = v;
    }
    if (threadIdx.x == 0) *flag = 1;
    __syncthreads();
    if (threadIdx.x < 32) {
        int c = threadIdx.x; float s = 0.0f;
        for (int e = 0; e < 512; e++) { float v = C[c * 512 + e]; s += v * v; }
        Csq[c] = s;
    }
}

// wave per row; lane = cluster
__global__ __launch_bounds__(256) void dkm_assign(
    const float* __restrict__ X, const float* __restrict__ Xsq,
    const float* __restrict__ Ct, const float* __restrict__ Csq,
    float* __restrict__ aout, const int* __restrict__ flag) {
    if (*flag == 0) return;
    int lane = threadIdx.x & 31, wave = threadIdx.x >> 5;
    int i = blockIdx.x * 8 + wave;
    const float* xr = X + (size_t)i * 512;
    float dot = 0.0f;
    for (int k = 0; k < 512; k++) dot += xr[k] * Ct[k * 32 + lane];
    float d2 = Xsq[i] + Csq[lane] - 2.0f * dot;
    float d  = sqrtf(fmaxf(d2, 0.0f));
    float lg = -d * 2.0f;                       // -d / TEMP, TEMP=0.5
    float mx = lg;
    for (int off = 16; off >= 1; off >>= 1) mx = fmaxf(mx, __shfl_xor(mx, off, 32));
    float e = __expf(lg - mx);
    float s = e;
    for (int off = 16; off >= 1; off >>= 1) s += __shfl_xor(s, off, 32);
    aout[(size_t)i * 32 + lane] = e / s;
}

__global__ __launch_bounds__(128) void dkm_num(
    const float* __restrict__ a, const float* __restrict__ X,
    float* __restrict__ num, const int* __restrict__ flag) {
    if (*flag == 0) return;
    int c = blockIdx.y;
    int e = blockIdx.x * 128 + threadIdx.x;
    float acc = 0.0f;
    for (int i = 0; i < 4096; i++)
        acc += a[(size_t)i * 32 + c] * X[(size_t)i * 512 + e];
    num[(size_t)c * 512 + e] = acc;
}

__global__ __launch_bounds__(256) void dkm_asum(
    const float* __restrict__ a, float* __restrict__ asum,
    const int* __restrict__ flag) {
    if (*flag == 0) return;
    __shared__ float red[256];
    int c = threadIdx.x & 31, part = threadIdx.x >> 5;
    float s = 0.0f;
    for (int i = part; i < 4096; i += 8) s += a[(size_t)i * 32 + c];
    red[threadIdx.x] = s;
    __syncthreads();
    if (part == 0) {
        float tt = 0.0f;
        for (int p = 0; p < 8; p++) tt += red[p * 32 + c];
        asum[c] = tt + 1e-6f;
    }
}

__global__ __launch_bounds__(256) void dkm_update(
    const float* __restrict__ num, const float* __restrict__ asum,
    float* __restrict__ C, int* __restrict__ flag, int is_last) {
    if (*flag == 0) return;
    __shared__ float red[256];
    __shared__ int contS;
    float ds = 0.0f;
    for (int p = threadIdx.x; p < 32 * 512; p += 256) {
        int c = p >> 9;
        ds += fabsf(num[p] / asum[c] - C[p]);
    }
    red[threadIdx.x] = ds;
    __syncthreads();
    for (int s = 128; s > 0; s >>= 1) {
        if (threadIdx.x < s) red[threadIdx.x] += red[threadIdx.x + s];
        __syncthreads();
    }
    if (threadIdx.x == 0) {
        int cont = red[0] > 1e-4f;
        contS = cont;
        if (!cont) *flag = 0;    // converged: this eval's C and a are final
    }
    __syncthreads();
    if (contS && !is_last) {
        for (int p = threadIdx.x; p < 32 * 512; p += 256) {
            int c = p >> 9;
            C[p] = num[p] / asum[c];
        }
    }
}

__global__ __launch_bounds__(256) void dkm_refresh(
    const float* __restrict__ C, float* __restrict__ Ct,
    float* __restrict__ Csq, const int* __restrict__ flag) {
    if (*flag == 0) return;
    for (int p = threadIdx.x; p < 32 * 512; p += 256) {
        int c = p >> 9, e = p & 511;
        Ct[e * 32 + c] = C[p];
    }
    if (threadIdx.x < 32) {
        int c = threadIdx.x; float s = 0.0f;
        for (int e = 0; e < 512; e++) { float v = C[c * 512 + e]; s += v * v; }
        Csq[c] = s;
    }
}

__global__ void copy_C(const float* __restrict__ C, float* __restrict__ out) {
    int p = blockIdx.x * 256 + threadIdx.x;
    if (p < 32 * 512) out[p] = C[p];
}

// ---------------------------------------------------------------------------
// launch
// ---------------------------------------------------------------------------
extern "C" void kernel_launch(void* const* d_in, const int* in_sizes, int n_in,
                              void* d_out, int out_size, void* d_ws, size_t ws_size,
                              hipStream_t stream)
{
    (void)in_sizes; (void)n_in; (void)out_size; (void)ws_size;
    const float* emb = (const float*)d_in[0];   // [4096,768]
    const float* qw  = (const float*)d_in[1];   // [768,512]
    const float* kw  = (const float*)d_in[2];
    const float* vw  = (const float*)d_in[3];
    const float* Wi  = (const float*)d_in[4];   // [1536,512]
    const float* bi  = (const float*)d_in[5];   // [1536]
    const float* Wo  = (const float*)d_in[6];   // [512,512]
    const float* bo  = (const float*)d_in[7];   // [512]
    const int*   idx = (const int*)d_in[8];     // [32]

    const size_t BIG = (size_t)4096 * 512;      // 2,097,152

    // ---- f32 region: X + DKM state ----
    float* ws   = (float*)d_ws;
    float* X    = ws;                        // tr, [4096,512]
    float* Xsq  = X + BIG;                   // 4096
    float* Ccur = Xsq + 4096;                // 32*512
    float* Ct   = Ccur + 32 * 512;           // 512*32
    float* Csq  = Ct + 32 * 512;             // 32
    float* num  = Csq + 32;                  // 32*512
    float* asum = num + 32 * 512;            // 32

    // ---- bf16 region (16B aligned: all counts are multiples of 8) ----
    unsigned short* Eb   = (unsigned short*)(asum + 32);      // [4096][768]
    unsigned short* qwT  = Eb  + (size_t)4096 * 768;          // [512][768]
    unsigned short* kwT  = qwT + (size_t)512 * 768;
    unsigned short* vwT  = kwT + (size_t)512 * 768;
    unsigned short* Wib  = vwT + (size_t)512 * 768;           // [1536][512]
    unsigned short* Wob  = Wib + (size_t)1536 * 512;          // [512][512]
    unsigned short* qact = Wob + (size_t)512 * 512;           // [4096][512]
    unsigned short* kact = qact + BIG;
    unsigned short* vact = kact + BIG;
    unsigned short* Qb   = vact + BIG;                        // [4096][512]
    unsigned short* Kb   = Qb + BIG;
    unsigned short* Vtb  = Kb + BIG;                          // [8][64][4096]
    unsigned short* Ob   = Vtb + BIG;                         // [4096][512]
    int* flag = (int*)(Ob + BIG);

    dim3 blk(256);
    dim3 gg(4096 / 128, 512 / 128);   // 32 x 4 tiles

    // ---- prepack to bf16 (weights transposed to uniform [N,K] layout) ----
    convert_bf16<<<(4096 * 768) / 256, blk, 0, stream>>>(emb, Eb, 4096 * 768);
    convert_bf16_T<<<(768 * 512) / 256, blk, 0, stream>>>(qw, qwT, 768, 512);
    convert_bf16_T<<<(768 * 512) / 256, blk, 0, stream>>>(kw, kwT, 768, 512);
    convert_bf16_T<<<(768 * 512) / 256, blk, 0, stream>>>(vw, vwT, 768, 512);
    convert_bf16<<<(1536 * 512) / 256, blk, 0, stream>>>(Wi, Wib, 1536 * 512);
    convert_bf16<<<(512 * 512) / 256, blk, 0, stream>>>(Wo, Wob, 512 * 512);

    // ---- q/k/v = E @ {qw,kw,vw}  (bf16 out, feeds next GEMM) ----
    gemm_bf16_async<<<gg, blk, 0, stream>>>(Eb, qwT, nullptr, nullptr, qact, 1, 4096, 512, 768);
    gemm_bf16_async<<<gg, blk, 0, stream>>>(Eb, kwT, nullptr, nullptr, kact, 1, 4096, 512, 768);
    gemm_bf16_async<<<gg, blk, 0, stream>>>(Eb, vwT, nullptr, nullptr, vact, 1, 4096, 512, 768);
    // ---- in-projection: x @ W.T + b ----
    gemm_bf16_async<<<gg, blk, 0, stream>>>(qact, Wib,                    bi,        nullptr, Qb,  1, 4096, 512, 512);
    gemm_bf16_async<<<gg, blk, 0, stream>>>(kact, Wib + (size_t)512*512,  bi + 512,  nullptr, Kb,  1, 4096, 512, 512);
    gemm_bf16_async<<<gg, blk, 0, stream>>>(vact, Wib + (size_t)1024*512, bi + 1024, nullptr, Vtb, 2, 4096, 512, 512);
    // ---- attention (bf16 out) ----
    flash_attn<<<dim3(32, 8), blk, 0, stream>>>(Qb, Kb, Vtb, Ob);
    // ---- output projection -> tr (X, f32) ----
    gemm_bf16_async<<<gg, blk, 0, stream>>>(Ob, Wob, bo, X, nullptr, 0, 4096, 512, 512);

    // ---- DKM ----
    xsq_kernel<<<16, blk, 0, stream>>>(X, Xsq);
    dkm_init<<<1, blk, 0, stream>>>(X, idx, Ccur, Ct, Csq, flag);

    float* aout = (float*)d_out + 32 * 512;   // d_out = [C (32x512) | a (4096x32)]
    for (int it = 0; it <= 100; it++) {
        dkm_assign<<<512, blk, 0, stream>>>(X, Xsq, Ct, Csq, aout, flag);
        dkm_num<<<dim3(4, 32), 128, 0, stream>>>(aout, X, num, flag);
        dkm_asum<<<1, blk, 0, stream>>>(aout, asum, flag);
        dkm_update<<<1, blk, 0, stream>>>(num, asum, Ccur, flag, it == 100 ? 1 : 0);
        dkm_refresh<<<1, blk, 0, stream>>>(Ccur, Ct, Csq, flag);
    }
    copy_C<<<64, blk, 0, stream>>>(Ccur, (float*)d_out);
}
